// PPOAgent_83038897701305
// MI455X (gfx1250) — matrix operand order repaired and checked by reading, not compiled, not run on hardware
//
#include <hip/hip_runtime.h>
#include <hip/hip_bf16.h>

// ---------------------------------------------------------------------------
// Problem constants (from reference)
// ---------------------------------------------------------------------------
#define TSEQ 1024
#define BSZ  512
#define INP  128
#define HID  256
#define ACT  3
#define KTOT (INP + HID)       // 384 fused GEMM K for LSTM
#define G4   (4 * HID)         // 1024 gate columns
#define NKT_L (KTOT / 32)      // 12 k-tiles (bf16 wmma K=32)
#define NCT_L 8                // col tiles per wave (8 waves * 8 * 16 = 1024)
#define ROWS_L 16              // batch rows per LSTM workgroup
#define HROWS 64               // rows per head workgroup
#define NROWS ((size_t)TSEQ * BSZ)

typedef __attribute__((ext_vector_type(16))) __bf16 v16bf;
typedef __attribute__((ext_vector_type(8)))  float  v8f;

__device__ __forceinline__ v8f wmma_bf16(v16bf a, v16bf b, v8f c) {
  return __builtin_amdgcn_wmma_f32_16x16x32_bf16(false, a, false, b,
                                                 (short)0, c, false, false);
}

__device__ __forceinline__ __hip_bfloat16 f2b(float f) { return __float2bfloat16(f); }
__device__ __forceinline__ float sigmoidf_fast(float x) { return 1.0f / (1.0f + __expf(-x)); }

// ---------------------------------------------------------------------------
// Workspace layout (bytes, all 1024-aligned)
// ---------------------------------------------------------------------------
#define OFF_WLSTM 0u                      // 64 gc * 12 kt * 1024B bf16 fragment blocks
#define OFF_BIAS  786432u                 // 1024 f32 (b_ih + b_hh)
#define OFF_WA1   790528u                 // 8ct*8kt*1024
#define OFF_WA2   856064u                 // 8ct*4kt*1024
#define OFF_WA3   888832u                 // 1ct*4kt*1024 (cols padded 3->16)
#define OFF_WV1   892928u
#define OFF_WV2   958464u
#define OFF_WV3   991232u
#define OFF_HID   995328u                 // NROWS*HID bf16 = 256 MiB

// d_out layout: logits | value | hn | cn
#define OUT_VALUE 1572864u
#define OUT_HN    2097152u
#define OUT_CN    2228224u

// ---------------------------------------------------------------------------
// Pack kernels: fp32 row-major weights -> bf16 wave32 B-fragment blocks.
// Block (ct,kt) = 1024 B: lane L holds col ct*16+L%16, 16 bf16 = K in ISA table order.
// ---------------------------------------------------------------------------
__global__ void pack_lstm_w(const float* __restrict__ wih,
                            const float* __restrict__ whh,
                            __hip_bfloat16* __restrict__ dst) {
  int idx = blockIdx.x * blockDim.x + threadIdx.x;   // 64*12*32*16 = 393216
  if (idx >= 64 * NKT_L * 512) return;
  int e    = idx & 15;
  int lane = (idx >> 4) & 31;
  int blk  = idx >> 9;
  int kt   = blk % NKT_L;
  int gc   = blk / NKT_L;
  int col  = gc * 16 + (lane & 15);          // gate column (0..1023)
  int k    = kt * 32 + (lane >> 4) * 8 + (e & 7) + (e >> 3) * 16;       // fused K (0..383)
  float v = (k < INP) ? wih[col * INP + k] : whh[col * HID + (k - INP)];
  dst[idx] = f2b(v);
}

__global__ void pack_b(const float* __restrict__ W, __hip_bfloat16* __restrict__ dst,
                       int ncols, int K, int nct, int nkt) {
  int idx = blockIdx.x * blockDim.x + threadIdx.x;
  if (idx >= nct * nkt * 512) return;
  int e    = idx & 15;
  int lane = (idx >> 4) & 31;
  int blk  = idx >> 9;
  int kt   = blk % nkt;
  int ct   = blk / nkt;
  int col  = ct * 16 + (lane & 15);
  int k    = kt * 32 + (lane >> 4) * 8 + (e & 7) + (e >> 3) * 16;
  float v = (col < ncols && k < K) ? W[col * K + k] : 0.0f;
  dst[idx] = f2b(v);
}

__global__ void pack_bias(const float* __restrict__ bih, const float* __restrict__ bhh,
                          float* __restrict__ b) {
  int i = blockIdx.x * blockDim.x + threadIdx.x;
  if (i < G4) b[i] = bih[i] + bhh[i];
}

// ---------------------------------------------------------------------------
// LSTM scan. Grid: 32 WGs x 256 thr (8 waves). WG owns 16 batch rows for all
// 1024 steps. Wave w owns hidden units [32w,32w+32): gate col-tiles
// {gate*16 + 2w, gate*16 + 2w + 1} so the full cell update stays in registers.
// W (bf16, 768KB) is streamed from L2 every step via GLOBAL b128 loads; an
// opaque per-iteration byte offset defeats LICM without breaking the global
// address-space inference (keeps loads off the FLAT/DScnt path).
// ---------------------------------------------------------------------------
__global__ __launch_bounds__(256) void lstm_scan_kernel(
    const float* __restrict__ x, const int* __restrict__ done,
    const float* __restrict__ h0, const float* __restrict__ c0,
    const __hip_bfloat16* __restrict__ wpack, const float* __restrict__ bias,
    __hip_bfloat16* __restrict__ hiddenb,
    float* __restrict__ out_hn, float* __restrict__ out_cn) {
  extern __shared__ char smem[];
  __hip_bfloat16* s_inp = (__hip_bfloat16*)smem;                 // 16 x 384 bf16 ([x | h])
  float*          s_c   = (float*)(smem + ROWS_L * KTOT * 2);    // 16 x 256 f32

  const int tid  = threadIdx.x;
  const int wave = tid >> 5;
  const int lane = tid & 31;
  const int r0   = blockIdx.x * ROWS_L;

  // init state from h0 / c0
  for (int i = tid; i < ROWS_L * HID; i += 256) {
    int m = i >> 8, j = i & (HID - 1);
    s_inp[m * KTOT + INP + j] = f2b(h0[(r0 + m) * HID + j]);
    s_c[m * HID + j] = c0[(r0 + m) * HID + j];
  }

  // preload gate biases for this wave's 32 units
  float bI[2], bF[2], bG[2], bO[2];
  {
    int nn = lane & 15;
    #pragma unroll
    for (int p = 0; p < 2; ++p) {
      int j = wave * 32 + p * 16 + nn;
      bI[p] = bias[j];
      bF[p] = bias[HID + j];
      bG[p] = bias[2 * HID + j];
      bO[p] = bias[3 * HID + j];
    }
  }

  const int m_a  = lane & 15;
  const int kb   = (lane >> 4) * 8;   // A-frag K sub-offset per ISA layout
  const int half = lane >> 4;
  const int n    = lane & 15;
  const v8f vzero = {};

  for (int t = 0; t < TSEQ; ++t) {
    // Opaque zero byte-offset, redefined every iteration: blocks LICM from
    // hoisting the streaming weight loads, while the base pointer stays
    // traceable to the kernarg => global_load_b128 (LOADcnt only, no FLAT).
    size_t zoff = 0;
    asm volatile("" : "+s"(zoff));
    const char* wp = (const char*)wpack + zoff;

    __syncthreads();
    // ---- phase 1: done-mask reset + stage x_t as bf16 (vectorized) ----
    {
      int row = tid >> 4;
      int sub = tid & 15;
      if (done[t * BSZ + r0 + row]) {
        const int j0 = sub * 16;
        uint4 z4 = {0u, 0u, 0u, 0u};
        *(uint4*)(s_inp + row * KTOT + INP + j0)     = z4;  // 8 bf16
        *(uint4*)(s_inp + row * KTOT + INP + j0 + 8) = z4;  // 8 bf16
        float4 zf = {0.f, 0.f, 0.f, 0.f};
        #pragma unroll
        for (int q = 0; q < 4; ++q)
          *(float4*)(s_c + row * HID + j0 + q * 4) = zf;
      }
      // x_t tile: 16x128 f32 -> bf16, float4 loads + b64 LDS stores
      const float4* xq = (const float4*)(x + (size_t)(t * BSZ + r0) * INP);
      for (int i = tid; i < ROWS_L * INP / 4; i += 256) {
        int m = i >> 5, kq = i & 31;           // 32 float4 per row
        float4 xv = xq[i];
        union { __hip_bfloat16 h[4]; uint2 u; } pk;
        pk.h[0] = f2b(xv.x); pk.h[1] = f2b(xv.y);
        pk.h[2] = f2b(xv.z); pk.h[3] = f2b(xv.w);
        *(uint2*)(s_inp + m * KTOT + kq * 4) = pk.u;
      }
    }
    __syncthreads();

    // prefetch next step's x tile (64 x 128B lines) + done flags into cache
    // while the GEMM runs (x is the only HBM-resident per-step stream)
    if (t + 1 < TSEQ) {
      const char* xnext = (const char*)(x + (size_t)((t + 1) * BSZ + r0) * INP);
      if (tid < 64) __builtin_prefetch(xnext + tid * 128, 0, 3);
      if (tid == 64) __builtin_prefetch(done + (t + 1) * BSZ + r0, 0, 3);
    }

    // ---- phase 2: gates = [x|h] @ W^T  (WMMA bf16, fp32 acc) ----
    v8f acc[NCT_L];
    #pragma unroll
    for (int ct = 0; ct < NCT_L; ++ct) acc[ct] = vzero;

    #pragma unroll
    for (int kt = 0; kt < NKT_L; ++kt) {
      const __hip_bfloat16* ap = s_inp + m_a * KTOT + kt * 32 + kb;
      union { uint4 q[2]; v16bf v; } A;
      A.q[0] = *(const uint4*)(ap);
      A.q[1] = *(const uint4*)(ap + 16);
      #pragma unroll
      for (int ct = 0; ct < NCT_L; ++ct) {
        const int gate = ct >> 1, p = ct & 1;
        const int gc = gate * 16 + wave * 2 + p;   // of 64 col tiles
        const char* bp = wp + (size_t)(gc * NKT_L + kt) * 1024 + lane * 32;
        union { uint4 q[2]; v16bf v; } B;
        B.q[0] = *(const uint4*)(bp);
        B.q[1] = *(const uint4*)(bp + 16);
        acc[ct] = wmma_bf16(A.v, B.v, acc[ct]);
      }
    }
    __syncthreads();

    // ---- phase 3: cell update fully in registers ----
    #pragma unroll
    for (int p = 0; p < 2; ++p) {
      const int j = wave * 32 + p * 16 + n;
      #pragma unroll
      for (int r = 0; r < 8; ++r) {
        const int m = r + half * 8;
        float gi = acc[0 + p][r] + bI[p];
        float gf = acc[2 + p][r] + bF[p];
        float gg = acc[4 + p][r] + bG[p];
        float go = acc[6 + p][r] + bO[p];
        float cprev = s_c[m * HID + j];
        float cnew = sigmoidf_fast(gf) * cprev + sigmoidf_fast(gi) * tanhf(gg);
        float hnew = sigmoidf_fast(go) * tanhf(cnew);
        s_c[m * HID + j] = cnew;
        s_inp[m * KTOT + INP + j] = f2b(hnew);
        hiddenb[((size_t)t * BSZ + r0 + m) * HID + j] = f2b(hnew);
        if (t == TSEQ - 1) {
          out_hn[(r0 + m) * HID + j] = hnew;
          out_cn[(r0 + m) * HID + j] = cnew;
        }
      }
    }
  }
}

// ---------------------------------------------------------------------------
// Head GEMM helper: 64x{K} @ {K}x128, tanh+bias epilogue -> LDS bf16.
// One "head half" = 4 waves; wave w4 covers col tiles {2*w4, 2*w4+1}, 4 row tiles.
// ---------------------------------------------------------------------------
__device__ __forceinline__ void head_gemm(
    const __hip_bfloat16* __restrict__ src, int sstride,
    const __hip_bfloat16* __restrict__ wp, int nkt,
    const float* __restrict__ bias,
    __hip_bfloat16* __restrict__ dst, int dstride,
    int w4, int lane) {
  const int m_a = lane & 15, kb = (lane >> 4) * 8;
  const int half = lane >> 4, n = lane & 15;
  const v8f vzero = {};
  v8f acc[4][2];
  #pragma unroll
  for (int rt = 0; rt < 4; ++rt)
    #pragma unroll
    for (int cp = 0; cp < 2; ++cp) acc[rt][cp] = vzero;

  for (int kt = 0; kt < nkt; ++kt) {
    union { uint4 q[2]; v16bf v; } A[4];
    #pragma unroll
    for (int rt = 0; rt < 4; ++rt) {
      const __hip_bfloat16* ap = src + (rt * 16 + m_a) * sstride + kt * 32 + kb;
      A[rt].q[0] = *(const uint4*)(ap);
      A[rt].q[1] = *(const uint4*)(ap + 16);
    }
    #pragma unroll
    for (int cp = 0; cp < 2; ++cp) {
      const int ctg = w4 * 2 + cp;
      const char* bp = (const char*)wp + (size_t)(ctg * nkt + kt) * 1024 + lane * 32;
      union { uint4 q[2]; v16bf v; } B;
      B.q[0] = *(const uint4*)(bp);
      B.q[1] = *(const uint4*)(bp + 16);
      #pragma unroll
      for (int rt = 0; rt < 4; ++rt)
        acc[rt][cp] = wmma_bf16(A[rt].v, B.v, acc[rt][cp]);
    }
  }
  #pragma unroll
  for (int cp = 0; cp < 2; ++cp) {
    const int col = (w4 * 2 + cp) * 16 + n;
    const float b = bias[col];
    #pragma unroll
    for (int rt = 0; rt < 4; ++rt)
      #pragma unroll
      for (int r = 0; r < 8; ++r) {
        const int m = rt * 16 + r + half * 8;
        dst[m * dstride + col] = f2b(tanhf(acc[rt][cp][r] + b));
      }
  }
}

// ---------------------------------------------------------------------------
// Heads: action (H->128->128->3, mask) and value (H->128->128->1).
// Grid: 8192 WGs x 256 thr; WG = 64 rows. Waves 0-3 action, 4-7 value.
// ---------------------------------------------------------------------------
__global__ __launch_bounds__(256) void heads_kernel(
    const __hip_bfloat16* __restrict__ hiddenb,
    const __hip_bfloat16* __restrict__ wa1, const __hip_bfloat16* __restrict__ wa2,
    const __hip_bfloat16* __restrict__ wa3,
    const __hip_bfloat16* __restrict__ wv1, const __hip_bfloat16* __restrict__ wv2,
    const __hip_bfloat16* __restrict__ wv3,
    const float* __restrict__ ba1, const float* __restrict__ ba2,
    const float* __restrict__ ba3,
    const float* __restrict__ bv1, const float* __restrict__ bv2,
    const float* __restrict__ bv3,
    const unsigned char* __restrict__ invalid,
    float* __restrict__ out_logits, float* __restrict__ out_value) {
  extern __shared__ char smem[];
  __hip_bfloat16* s_h  = (__hip_bfloat16*)smem;                       // 64x256, reused as m2 (2x 64x128)
  __hip_bfloat16* s_m1 = (__hip_bfloat16*)(smem + HROWS * HID * 2);   // 2x 64x128 (action|value)

  const int tid = threadIdx.x, wave = tid >> 5, lane = tid & 31;
  const size_t row0 = (size_t)blockIdx.x * HROWS;

  { // stage hidden tile (bf16, 128-bit loads)
    const uint4* src = (const uint4*)(hiddenb + row0 * HID);
    uint4* dst = (uint4*)s_h;
    for (int i = tid; i < HROWS * HID / 8; i += 256) dst[i] = src[i];
  }
  __syncthreads();

  __hip_bfloat16* m1a = s_m1;
  __hip_bfloat16* m1v = s_m1 + HROWS * 128;

  // layer 1: 64x256 -> 64x128, tanh
  if (wave < 4) head_gemm(s_h, HID, wa1, 8, ba1, m1a, 128, wave, lane);
  else          head_gemm(s_h, HID, wv1, 8, bv1, m1v, 128, wave - 4, lane);
  __syncthreads();

  __hip_bfloat16* m2a = s_h;               // reuse hidden-tile space
  __hip_bfloat16* m2v = s_h + HROWS * 128;

  // layer 2: 64x128 -> 64x128, tanh
  if (wave < 4) head_gemm(m1a, 128, wa2, 4, ba2, m2a, 128, wave, lane);
  else          head_gemm(m1v, 128, wv2, 4, bv2, m2v, 128, wave - 4, lane);
  __syncthreads();

  // layer 3: 64x128 -> 64x{3|1} via zero-padded 16-col WMMA tile
  {
    const int m_a = lane & 15, kb = (lane >> 4) * 8;
    const int half = lane >> 4, n = lane & 15;
    const v8f vzero = {};
    const int rt = (wave < 4) ? wave : (wave - 4);
    const __hip_bfloat16* src = (wave < 4) ? m2a : m2v;
    const __hip_bfloat16* wp  = (wave < 4) ? wa3 : wv3;
    v8f acc = vzero;
    #pragma unroll
    for (int kt = 0; kt < 4; ++kt) {
      const __hip_bfloat16* ap = src + (rt * 16 + m_a) * 128 + kt * 32 + kb;
      union { uint4 q[2]; v16bf v; } A;
      A.q[0] = *(const uint4*)(ap);
      A.q[1] = *(const uint4*)(ap + 16);
      const char* bp = (const char*)wp + (size_t)kt * 1024 + lane * 32;
      union { uint4 q[2]; v16bf v; } B;
      B.q[0] = *(const uint4*)(bp);
      B.q[1] = *(const uint4*)(bp + 16);
      acc = wmma_bf16(A.v, B.v, acc);
    }
    if (wave < 4) {
      if (n < ACT) {
        const float b = ba3[n];
        #pragma unroll
        for (int r = 0; r < 8; ++r) {
          const size_t grow = row0 + rt * 16 + r + half * 8;
          float v = acc[r] + b;
          if (invalid[grow * ACT + n]) v = -1.0e8f;   // CategoricalMasked
          out_logits[grow * ACT + n] = v;
        }
      }
    } else {
      if (n == 0) {
        const float b = bv3[0];
        #pragma unroll
        for (int r = 0; r < 8; ++r) {
          const size_t grow = row0 + rt * 16 + r + half * 8;
          out_value[grow] = acc[r] + b;
        }
      }
    }
  }
}

// ---------------------------------------------------------------------------
extern "C" void kernel_launch(void* const* d_in, const int* in_sizes, int n_in,
                              void* d_out, int out_size, void* d_ws, size_t ws_size,
                              hipStream_t stream) {
  const float* x    = (const float*)d_in[0];
  const int*   done = (const int*)d_in[1];
  const unsigned char* invalid = (const unsigned char*)d_in[2];  // bool mask
  const float* h0   = (const float*)d_in[3];
  const float* c0   = (const float*)d_in[4];
  const float* Wih  = (const float*)d_in[5];
  const float* Whh  = (const float*)d_in[6];
  const float* bih  = (const float*)d_in[7];
  const float* bhh  = (const float*)d_in[8];
  const float* Wa1  = (const float*)d_in[9];
  const float* ba1  = (const float*)d_in[10];
  const float* Wa2  = (const float*)d_in[11];
  const float* ba2  = (const float*)d_in[12];
  const float* Wa3  = (const float*)d_in[13];
  const float* ba3  = (const float*)d_in[14];
  const float* Wv1  = (const float*)d_in[15];
  const float* bv1  = (const float*)d_in[16];
  const float* Wv2  = (const float*)d_in[17];
  const float* bv2  = (const float*)d_in[18];
  const float* Wv3  = (const float*)d_in[19];
  const float* bv3  = (const float*)d_in[20];

  char* ws = (char*)d_ws;
  __hip_bfloat16* wpack = (__hip_bfloat16*)(ws + OFF_WLSTM);
  float*          biasw = (float*)(ws + OFF_BIAS);
  __hip_bfloat16* wa1p  = (__hip_bfloat16*)(ws + OFF_WA1);
  __hip_bfloat16* wa2p  = (__hip_bfloat16*)(ws + OFF_WA2);
  __hip_bfloat16* wa3p  = (__hip_bfloat16*)(ws + OFF_WA3);
  __hip_bfloat16* wv1p  = (__hip_bfloat16*)(ws + OFF_WV1);
  __hip_bfloat16* wv2p  = (__hip_bfloat16*)(ws + OFF_WV2);
  __hip_bfloat16* wv3p  = (__hip_bfloat16*)(ws + OFF_WV3);
  __hip_bfloat16* hidb  = (__hip_bfloat16*)(ws + OFF_HID);

  float* out = (float*)d_out;

  // 1) pack weights into wave32 bf16 B-fragment blocks
  pack_lstm_w<<<(64 * NKT_L * 512 + 255) / 256, 256, 0, stream>>>(Wih, Whh, wpack);
  pack_bias<<<(G4 + 255) / 256, 256, 0, stream>>>(bih, bhh, biasw);
  pack_b<<<(8 * 8 * 512 + 255) / 256, 256, 0, stream>>>(Wa1, wa1p, 128, HID, 8, 8);
  pack_b<<<(8 * 4 * 512 + 255) / 256, 256, 0, stream>>>(Wa2, wa2p, 128, 128, 8, 4);
  pack_b<<<(1 * 4 * 512 + 255) / 256, 256, 0, stream>>>(Wa3, wa3p, ACT, 128, 1, 4);
  pack_b<<<(8 * 8 * 512 + 255) / 256, 256, 0, stream>>>(Wv1, wv1p, 128, HID, 8, 8);
  pack_b<<<(8 * 4 * 512 + 255) / 256, 256, 0, stream>>>(Wv2, wv2p, 128, 128, 8, 4);
  pack_b<<<(1 * 4 * 512 + 255) / 256, 256, 0, stream>>>(Wv3, wv3p, 1, 128, 1, 4);

  // 2) sequential LSTM scan: 32 independent batch-slice WGs, WMMA per step
  const size_t lds_lstm = (size_t)ROWS_L * KTOT * 2 + (size_t)ROWS_L * HID * 4; // 28672
  lstm_scan_kernel<<<BSZ / ROWS_L, 256, lds_lstm, stream>>>(
      x, done, h0, c0, wpack, biasw, hidb, out + OUT_HN, out + OUT_CN);

  // 3) heads over all T*BS rows
  const size_t lds_heads = (size_t)HROWS * HID * 2 + (size_t)2 * HROWS * 128 * 2; // 65536
  heads_kernel<<<(int)(NROWS / HROWS), 256, lds_heads, stream>>>(
      hidb, wa1p, wa2p, wa3p, wv1p, wv2p, wv3p,
      ba1, ba2, ba3, bv1, bv2, bv3, invalid, out, out + OUT_VALUE);
}
